// NonRigidDeform_12463995093362
// MI455X (gfx1250) — matrix operand order repaired and checked by reading, not compiled, not run on hardware
//
#include <hip/hip_runtime.h>
#include <hip/hip_bf16.h>
#include <math.h>

// ---- CDNA5 WMMA types (wave32) ----
typedef __attribute__((ext_vector_type(16))) __bf16 v16bf;
typedef __attribute__((ext_vector_type(8)))  float  v8f;

#define KC 80      // clusters
#define FD 64      // feature dim
#define JD 24      // joints
#define PD 64      // pose feature dim
#define HD 128
#define VMAX 6890
#define EPSF 1e-6f
#define INV_SIGMA 10.0f   // 1/0.1

// workspace layout (floats)
#define WS_CNT    0        // 80
#define WS_JXYZS  80       // 240  (a^T @ xyz accumulators)
#define WS_JFEATS 320      // 5120 (a^T @ feature accumulators)
#define WS_JXYZ   5440     // 240  (final joint_xyz)
#define WS_Q      5680     // 320
#define WS_T      6000     // 240
#define WS_ACCUM_FLOATS 5440

// ================================================================
// Kernel A: per-point cluster softmax (WMMA GEMM1: feature @ centers^T)
// + block-local reduction of cnt / a^T@xyz / a^T@feature (WMMA GEMM2).
// 128 threads = 4 wave32; each slab = 64 points.
// ================================================================
__launch_bounds__(128)
__global__ void kA(const float* __restrict__ xyz,
                   const float* __restrict__ feature,
                   const float* __restrict__ centers,
                   float* __restrict__ ws, int N)
{
    __shared__ float s_logits[64 * 85];   // pitch 85 (odd vs 64 banks)
    __shared__ float s_part[128];
    __shared__ float s_rnorm[64];
    __shared__ float s_cnorm[KC];
    __shared__ float s_xyz[64 * 3];

    const int tid  = threadIdx.x;
    const int w    = tid >> 5;        // wave id (wave32)
    const int lane = tid & 31;
    const int half = lane >> 4;       // lane group 0..15 / 16..31
    const int li   = lane & 15;

    // persistent split-K accumulators for a^T @ feature (5 cluster M-tiles)
    v8f accJF[5];
#pragma unroll
    for (int t = 0; t < 5; ++t) accJF[t] = (v8f){0,0,0,0,0,0,0,0};
    float c_s = 0.f, c_x = 0.f, c_y = 0.f, c_z = 0.f;

    // cluster norms (once): |centers[k]|^2
    if (tid < KC) {
        const float* cr = centers + tid * FD;
        float s = 0.f;
#pragma unroll 8
        for (int j = 0; j < FD; ++j) { float v = cr[j]; s += v * v; }
        s_cnorm[tid] = s;
    }

    const int nSlab = (N + 63) >> 6;
    for (int slab = blockIdx.x; slab < nSlab; slab += gridDim.x) {
        const int p0 = slab << 6;

        // stage this slab's xyz
        if (tid < 64) {
            int p = p0 + tid;
            float x = 0.f, y = 0.f, z = 0.f;
            if (p < N) { x = xyz[p*3+0]; y = xyz[p*3+1]; z = xyz[p*3+2]; }
            s_xyz[tid*3+0] = x; s_xyz[tid*3+1] = y; s_xyz[tid*3+2] = z;
        }

        // ---- A fragments (feature rows, 16x32 bf16 layout) + row-norm partials
        const int row = p0 + w*16 + li;
        v16bf A0, A1;
        float pn = 0.f;
        {
            float vals[32];
            if (row < N) {
                const float* fr = feature + (size_t)row * FD;
#pragma unroll
                for (int c = 0; c < 2; ++c) {
                    const float4* q0 = (const float4*)(fr + c*32 + 16*half);
                    float4 x0 = q0[0], x1 = q0[1];
                    const float4* q1 = (const float4*)(fr + c*32 + 16*half + 8);
                    float4 y0 = q1[0], y1 = q1[1];
                    float* d = vals + c*16;
                    d[0]=x0.x; d[1]=x0.y; d[2]=x0.z; d[3]=x0.w;
                    d[4]=x1.x; d[5]=x1.y; d[6]=x1.z; d[7]=x1.w;
                    d[8]=y0.x; d[9]=y0.y; d[10]=y0.z; d[11]=y0.w;
                    d[12]=y1.x; d[13]=y1.y; d[14]=y1.z; d[15]=y1.w;
                }
            } else {
#pragma unroll
                for (int i = 0; i < 32; ++i) vals[i] = 0.f;
            }
#pragma unroll
            for (int i = 0; i < 32; ++i) pn += vals[i] * vals[i];
#pragma unroll
            for (int i = 0; i < 16; ++i) A0[i] = (__bf16)vals[i];
#pragma unroll
            for (int i = 0; i < 16; ++i) A1[i] = (__bf16)vals[16 + i];
        }
        s_part[tid] = pn;
        __syncthreads();
        if (tid < 64) {
            int ww = tid >> 4, ii = tid & 15;
            s_rnorm[tid] = s_part[ww*32 + ii] + s_part[ww*32 + 16 + ii];
        }
        __syncthreads();

        // ---- GEMM1: logits tiles (5 cluster tiles, K=64 -> 2 WMMAs each)
#pragma unroll
        for (int t = 0; t < 5; ++t) {
            v16bf B0f, B1f;
            const float* cb = centers + (t*16 + li) * FD + 16*half;
#pragma unroll
            for (int j = 0; j < 16; ++j) B0f[j] = (__bf16)cb[j];
#pragma unroll
            for (int j = 0; j < 16; ++j) B1f[j] = (__bf16)cb[32 + j];
            v8f acc = (v8f){0,0,0,0,0,0,0,0};
            acc = __builtin_amdgcn_wmma_f32_16x16x32_bf16(false, A0, false, B0f,
                                                          (short)0, acc, false, false);
            acc = __builtin_amdgcn_wmma_f32_16x16x32_bf16(false, A1, false, B1f,
                                                          (short)0, acc, false, false);
            const float cn = s_cnorm[t*16 + li];
#pragma unroll
            for (int r = 0; r < 8; ++r) {
                int m = w*16 + half*8 + r;                 // local point row
                s_logits[m*85 + t*16 + li] = 2.0f*acc[r] - s_rnorm[m] - cn;
            }
        }
        __syncthreads();

        // ---- per-point softmax over 80 clusters (in place -> a values)
        if (tid < 64) {
            float* rowp = s_logits + tid * 85;
            const bool valid = (p0 + tid) < N;
            float mx = -1e30f;
            for (int k = 0; k < KC; ++k) mx = fmaxf(mx, rowp[k]);
            float sum = 0.f;
            for (int k = 0; k < KC; ++k) { float e = __expf(rowp[k] - mx); rowp[k] = e; sum += e; }
            const float inv = valid ? (1.0f / sum) : 0.0f;
            for (int k = 0; k < KC; ++k) rowp[k] *= inv;
        }
        __syncthreads();

        // ---- cnt & a^T@xyz accumulation (register-resident across slabs)
        if (tid < KC) {
            for (int p = 0; p < 64; ++p) {
                float a = s_logits[p*85 + tid];
                c_s += a;
                c_x += a * s_xyz[p*3+0];
                c_y += a * s_xyz[p*3+1];
                c_z += a * s_xyz[p*3+2];
            }
        }

        // ---- GEMM2: accJF += a^T(80x64pts) @ feature(64pts x 64F)
        // wave w owns feature N-tile w; B frags reused across 5 M-tiles.
        v16bf Bf0, Bf1;
#pragma unroll
        for (int j = 0; j < 16; ++j) {
            int pA = p0 + 16*half + j;         // frag0: points 0..31
            int pB = p0 + 32 + 16*half + j;    // frag1: points 32..63
            float vA = (pA < N) ? feature[(size_t)pA * FD + w*16 + li] : 0.f;
            float vB = (pB < N) ? feature[(size_t)pB * FD + w*16 + li] : 0.f;
            Bf0[j] = (__bf16)vA;
            Bf1[j] = (__bf16)vB;
        }
#pragma unroll
        for (int mt = 0; mt < 5; ++mt) {
            v16bf Aa0, Aa1;
#pragma unroll
            for (int j = 0; j < 8; ++j) {
                int pA = 16*half + j;        // K = 16*half + j
                int pB = 16*half + 8 + j;    // K = 16*half + 8 + j
                Aa0[j]     = (__bf16)s_logits[pA*85        + mt*16 + li];
                Aa0[8 + j] = (__bf16)s_logits[pB*85        + mt*16 + li];
                Aa1[j]     = (__bf16)s_logits[(32 + pA)*85 + mt*16 + li];
                Aa1[8 + j] = (__bf16)s_logits[(32 + pB)*85 + mt*16 + li];
            }
            accJF[mt] = __builtin_amdgcn_wmma_f32_16x16x32_bf16(false, Aa0, false, Bf0,
                                                                (short)0, accJF[mt], false, false);
            accJF[mt] = __builtin_amdgcn_wmma_f32_16x16x32_bf16(false, Aa1, false, Bf1,
                                                                (short)0, accJF[mt], false, false);
        }
        __syncthreads();   // protect s_logits/s_xyz before next slab
    }

    // ---- flush split-K partials
    if (tid < KC) {
        atomicAdd(&ws[WS_CNT + tid], c_s);
        atomicAdd(&ws[WS_JXYZS + tid*3+0], c_x);
        atomicAdd(&ws[WS_JXYZS + tid*3+1], c_y);
        atomicAdd(&ws[WS_JXYZS + tid*3+2], c_z);
    }
#pragma unroll
    for (int mt = 0; mt < 5; ++mt) {
#pragma unroll
        for (int r = 0; r < 8; ++r) {
            int kk = mt*16 + half*8 + r;          // cluster
            atomicAdd(&ws[WS_JFEATS + kk*FD + w*16 + li], accJF[mt][r]);
        }
    }
}

// ================================================================
// Kernel B: joint-level pipeline (K=80). Single block; verts staged
// in CDNA5's large LDS (82KB of the 320KB pool, reused for h).
// ================================================================
__launch_bounds__(256)
__global__ void kB(const float* __restrict__ smpl_verts,
                   const float* __restrict__ skinning_weights,
                   const float* __restrict__ body_pose_feature,
                   const float* __restrict__ w_body,  const float* __restrict__ b_body,
                   const float* __restrict__ w_pos1,  const float* __restrict__ b_pos1,
                   const float* __restrict__ w_pos2,  const float* __restrict__ b_pos2,
                   const float* __restrict__ w_rot,   const float* __restrict__ b_rot,
                   const float* __restrict__ w_trans, const float* __restrict__ b_trans,
                   float* __restrict__ ws, int V)
{
    __shared__ float s_verts[VMAX * 3];   // later reused as h[80*128]
    __shared__ float s_gcat[KC * 128];    // [body_feat | joint_feat]; later g2
    __shared__ float s_g[KC * 64];
    __shared__ float s_jxyz[KC * 3];
    __shared__ float s_lohi[6];
    __shared__ float s_red[256 * 6];

    const int tid = threadIdx.x;

    // finalize joint_xyz and joint_feat (divide by cnt+EPS)
    for (int o = tid; o < KC*3; o += 256) {
        int k = o / 3;
        float c = ws[WS_CNT + k] + EPSF;
        float v = ws[WS_JXYZS + o] / c;
        s_jxyz[o] = v;
        ws[WS_JXYZ + o] = v;
    }
    for (int o = tid; o < KC*FD; o += 256) {
        int k = o >> 6, f = o & 63;
        float c = ws[WS_CNT + k] + EPSF;
        s_gcat[k*128 + 64 + f] = ws[WS_JFEATS + o] / c;
    }

    // stage verts + min/max reduction
    float lo0 = 1e30f, lo1 = 1e30f, lo2 = 1e30f;
    float hi0 = -1e30f, hi1 = -1e30f, hi2 = -1e30f;
    for (int v = tid; v < V; v += 256) {
        float x = smpl_verts[v*3+0], y = smpl_verts[v*3+1], z = smpl_verts[v*3+2];
        s_verts[v*3+0] = x; s_verts[v*3+1] = y; s_verts[v*3+2] = z;
        lo0 = fminf(lo0, x); lo1 = fminf(lo1, y); lo2 = fminf(lo2, z);
        hi0 = fmaxf(hi0, x); hi1 = fmaxf(hi1, y); hi2 = fmaxf(hi2, z);
    }
    s_red[tid*6+0] = lo0; s_red[tid*6+1] = lo1; s_red[tid*6+2] = lo2;
    s_red[tid*6+3] = hi0; s_red[tid*6+4] = hi1; s_red[tid*6+5] = hi2;
    __syncthreads();
    for (int s = 128; s > 0; s >>= 1) {
        if (tid < s) {
            for (int d = 0; d < 3; ++d) {
                s_red[tid*6+d]   = fminf(s_red[tid*6+d],   s_red[(tid+s)*6+d]);
                s_red[tid*6+3+d] = fmaxf(s_red[tid*6+3+d], s_red[(tid+s)*6+3+d]);
            }
        }
        __syncthreads();
    }
    if (tid < 6) s_lohi[tid] = s_red[tid];
    __syncthreads();

    // argmin over verts + body_feat = skinning_weights[idx] @ body_pose_feature
    if (tid < KC) {
        float jx = s_jxyz[tid*3+0], jy = s_jxyz[tid*3+1], jz = s_jxyz[tid*3+2];
        float best = 1e30f; int bi = 0;
        for (int v = 0; v < V; ++v) {
            float dx = jx - s_verts[v*3+0];
            float dy = jy - s_verts[v*3+1];
            float dz = jz - s_verts[v*3+2];
            float d = dx*dx + dy*dy + dz*dz;
            if (d < best) { best = d; bi = v; }
        }
        const float* W = skinning_weights + (size_t)bi * JD;
        for (int p = 0; p < PD; ++p) {
            float s = 0.f;
            for (int j = 0; j < JD; ++j) s += W[j] * body_pose_feature[j*PD + p];
            s_gcat[tid*128 + p] = s;
        }
    }
    __syncthreads();

    // g = gelu(gcat @ w_body + b_body)   [80 x 64], K=128
    for (int o = tid; o < KC*64; o += 256) {
        int k = o >> 6, m = o & 63;
        float s = b_body[m];
        const float* gr = s_gcat + k*128;
        for (int j = 0; j < 128; ++j) s += gr[j] * w_body[j*64 + m];
        float x3 = s * s * s;
        float tin = 0.7978845608028654f * (s + 0.044715f * x3);
        s_g[o] = 0.5f * s * (1.0f + tanhf(tin));
    }
    __syncthreads();

    // h = relu([xn, g] @ w_pos1 + b_pos1)  [80 x 128], K=67 ; reuse s_verts
    float* s_h = s_verts;
    for (int o = tid; o < KC*HD; o += 256) {
        int k = o >> 7, u = o & 127;
        float s = b_pos1[u];
        for (int d = 0; d < 3; ++d) {
            float lo = s_lohi[d], hi = s_lohi[3+d];
            float xn = 2.0f * (s_jxyz[k*3+d] - lo) / (hi - lo + EPSF) - 1.0f;
            s += xn * w_pos1[d*HD + u];
        }
        const float* gr = s_g + k*64;
        for (int m = 0; m < 64; ++m) s += gr[m] * w_pos1[(3+m)*HD + u];
        s_h[o] = fmaxf(s, 0.0f);
    }
    __syncthreads();

    // g2 = h @ w_pos2 + b_pos2   [80 x 64], K=128 ; reuse s_gcat
    float* s_g2 = s_gcat;
    for (int o = tid; o < KC*64; o += 256) {
        int k = o >> 6, m = o & 63;
        float s = b_pos2[m];
        const float* hr = s_h + k*HD;
        for (int u = 0; u < HD; ++u) s += hr[u] * w_pos2[u*64 + m];
        s_g2[k*64 + m] = s;
    }
    __syncthreads();

    // q (normalized) and t per joint
    if (tid < KC) {
        const float* g2 = s_g2 + tid*64;
        float q[4];
        for (int c = 0; c < 4; ++c) {
            float s = b_rot[c];
            for (int m = 0; m < 64; ++m) s += g2[m] * w_rot[m*4 + c];
            q[c] = s;
        }
        float n = sqrtf(q[0]*q[0] + q[1]*q[1] + q[2]*q[2] + q[3]*q[3]) + EPSF;
        for (int c = 0; c < 4; ++c) ws[WS_Q + tid*4 + c] = q[c] / n;
        for (int c = 0; c < 3; ++c) {
            float s = b_trans[c];
            for (int m = 0; m < 64; ++m) s += g2[m] * w_trans[m*3 + c];
            ws[WS_T + tid*3 + c] = s;
        }
    }
}

// ================================================================
// Kernel C: per-point 80-way softmax blend + quaternion rotate.
// ================================================================
__launch_bounds__(256)
__global__ void kC(const float* __restrict__ xyz,
                   const float* __restrict__ ws,
                   float* __restrict__ out, int N)
{
    __shared__ float s_jx[KC*3];
    __shared__ float s_q[KC*4];
    __shared__ float s_t[KC*3];
    const int tid = threadIdx.x;
    for (int o = tid; o < KC*3; o += 256) s_jx[o] = ws[WS_JXYZ + o];
    for (int o = tid; o < KC*4; o += 256) s_q[o]  = ws[WS_Q + o];
    for (int o = tid; o < KC*3; o += 256) s_t[o]  = ws[WS_T + o];
    __syncthreads();

    const int p = blockIdx.x * 256 + tid;
    if (p >= N) return;
    const float x = xyz[p*3+0], y = xyz[p*3+1], z = xyz[p*3+2];

    float mx = -1e30f;
    for (int k = 0; k < KC; ++k) {
        float dx = x - s_jx[k*3+0], dy = y - s_jx[k*3+1], dz = z - s_jx[k*3+2];
        mx = fmaxf(mx, -(dx*dx + dy*dy + dz*dz) * INV_SIGMA);
    }
    float sum = 0.f;
    float qb0 = 0.f, qb1 = 0.f, qb2 = 0.f, qb3 = 0.f;
    float tb0 = 0.f, tb1 = 0.f, tb2 = 0.f;
    for (int k = 0; k < KC; ++k) {
        float dx = x - s_jx[k*3+0], dy = y - s_jx[k*3+1], dz = z - s_jx[k*3+2];
        float l = -(dx*dx + dy*dy + dz*dz) * INV_SIGMA;
        float e = __expf(l - mx);
        sum += e;
        qb0 += e * s_q[k*4+0]; qb1 += e * s_q[k*4+1];
        qb2 += e * s_q[k*4+2]; qb3 += e * s_q[k*4+3];
        tb0 += e * s_t[k*3+0]; tb1 += e * s_t[k*3+1]; tb2 += e * s_t[k*3+2];
    }
    const float inv = 1.0f / sum;
    qb0 *= inv; qb1 *= inv; qb2 *= inv; qb3 *= inv;
    tb0 *= inv; tb1 *= inv; tb2 *= inv;
    const float qn = sqrtf(qb0*qb0 + qb1*qb1 + qb2*qb2 + qb3*qb3) + EPSF;
    const float qw = qb0 / qn, qx = qb1 / qn, qy = qb2 / qn, qz = qb3 / qn;

    // t2 = 2*cross(qv, v); out = v + qw*t2 + cross(qv, t2) + w@t
    const float tx = 2.0f * (qy*z - qz*y);
    const float ty = 2.0f * (qz*x - qx*z);
    const float tz = 2.0f * (qx*y - qy*x);
    out[p*3+0] = x + qw*tx + (qy*tz - qz*ty) + tb0;
    out[p*3+1] = y + qw*ty + (qz*tx - qx*tz) + tb1;
    out[p*3+2] = z + qw*tz + (qx*ty - qy*tx) + tb2;
}

// ================================================================
extern "C" void kernel_launch(void* const* d_in, const int* in_sizes, int n_in,
                              void* d_out, int out_size, void* d_ws, size_t ws_size,
                              hipStream_t stream)
{
    (void)n_in; (void)out_size; (void)ws_size;
    const float* xyz               = (const float*)d_in[0];
    const float* feature           = (const float*)d_in[1];
    const float* centers           = (const float*)d_in[2];
    const float* smpl_verts        = (const float*)d_in[3];
    const float* skinning_weights  = (const float*)d_in[4];
    const float* body_pose_feature = (const float*)d_in[5];
    const float* w_body  = (const float*)d_in[6];
    const float* b_body  = (const float*)d_in[7];
    const float* w_pos1  = (const float*)d_in[8];
    const float* b_pos1  = (const float*)d_in[9];
    const float* w_pos2  = (const float*)d_in[10];
    const float* b_pos2  = (const float*)d_in[11];
    const float* w_rot   = (const float*)d_in[12];
    const float* b_rot   = (const float*)d_in[13];
    const float* w_trans = (const float*)d_in[14];
    const float* b_trans = (const float*)d_in[15];

    const int N = in_sizes[0] / 3;
    const int V = in_sizes[3] / 3;
    float* ws = (float*)d_ws;

    // zero split-K accumulators every launch (graph-capture safe)
    hipMemsetAsync(ws, 0, (size_t)WS_ACCUM_FLOATS * sizeof(float), stream);

    int nSlab = (N + 63) / 64;
    int gridA = nSlab < 1024 ? nSlab : 1024;
    kA<<<gridA, 128, 0, stream>>>(xyz, feature, centers, ws, N);
    kB<<<1, 256, 0, stream>>>(smpl_verts, skinning_weights, body_pose_feature,
                              w_body, b_body, w_pos1, b_pos1, w_pos2, b_pos2,
                              w_rot, b_rot, w_trans, b_trans, ws, V);
    kC<<<(N + 255) / 256, 256, 0, stream>>>(xyz, ws, (float*)d_out, N);
}
